// HNLoRALinear_64183991271648
// MI455X (gfx1250) — matrix operand order repaired and checked by reading, not compiled, not run on hardware
//
#include <hip/hip_runtime.h>
#include <hip/hip_bf16.h>

// ---------------------------------------------------------------------------
// LoRA linear: out = x @ W^T + bias + 2.0 * (x @ A_b) @ B_b
// B=4, S=2048, D_IN=D_OUT=4096, R=16.  f32 in/out; main GEMM in bf16 WMMA
// (v_wmma_f32_16x16x32_bf16, f32 accumulate) with async global->LDS staging.
// Block tile 128x256, 8 wave32 (2x4), wave tile 64x64 = 4x4 WMMA tiles.
// ---------------------------------------------------------------------------

typedef __attribute__((ext_vector_type(16))) __bf16 v16bf;
typedef __attribute__((ext_vector_type(8)))  float  v8f;

#define M_TOTAL 8192
#define N_TOTAL 4096
#define K_TOTAL 4096
#define BM 128
#define BN 256
#define BK 32
#define LDS_ROW_BYTES 80                    // 64B data + 16B pad (conflict-free b128)
#define AT_BYTES (128 * LDS_ROW_BYTES)      // 10240
#define BT_BYTES (256 * LDS_ROW_BYTES)      // 20480
#define BUF_BYTES (AT_BYTES + BT_BYTES)     // 30720
#define SCALE 2.0f                          // LORA_ALPHA / LORA_RANK

union FragBf { v16bf v; uint4 q[2]; };

__device__ __forceinline__ void async_copy_b128(unsigned lds_off, unsigned long long gaddr) {
    asm volatile("global_load_async_to_lds_b128 %0, %1, off"
                 :: "v"(lds_off), "v"(gaddr) : "memory");
}
__device__ __forceinline__ void wait_async0() {
    asm volatile("s_wait_asynccnt 0" ::: "memory");
}

// ---------------- f32 -> bf16 conversion (grid-stride) ----------------------
__global__ void cvt_f32_to_bf16(const float* __restrict__ src,
                                __bf16* __restrict__ dst, int n) {
    int i = blockIdx.x * blockDim.x + threadIdx.x;
    int stride = gridDim.x * blockDim.x;
    for (; i < n; i += stride) dst[i] = (__bf16)src[i];
}

// ---------------- low = x @ lora_A (per batch), f32 -------------------------
__global__ void lora_low_kernel(const float* __restrict__ x,
                                const float* __restrict__ loraA,
                                float* __restrict__ low) {
    int t = blockIdx.x * blockDim.x + threadIdx.x;   // 0 .. 131071
    int r = t & 15;
    int row = t >> 4;                                 // 0 .. 8191
    int b = row >> 11;                                // row / 2048
    const float* xr = x + (size_t)row * K_TOTAL;
    const float* Ab = loraA + (size_t)b * K_TOTAL * 16 + r;
    float s = 0.0f;
#pragma unroll 8
    for (int k = 0; k < K_TOTAL; ++k)
        s = fmaf(xr[k], Ab[(size_t)k * 16], s);
    low[t] = s;
}

// ---------------- main fused GEMM + bias + LoRA-B epilogue ------------------
// grid = (N/BN, M/BM) = (16, 64); block = 256 threads = 8 wave32 (2x4 grid).
__global__ void __launch_bounds__(256)
lora_gemm_kernel(const __bf16* __restrict__ xb,   // [8192][4096] bf16
                 const __bf16* __restrict__ wb,   // [4096][4096] bf16 ([n][k])
                 const float*  __restrict__ bias, // [4096]
                 const float*  __restrict__ low,  // [8192][16]
                 const float*  __restrict__ loraB,// [4][16][4096]
                 float*        __restrict__ out)  // [8192][4096]
{
    __shared__ __align__(1024) unsigned char smem[2 * BUF_BYTES]; // [A0 B0][A1 B1]

    const int tid  = threadIdx.x;
    const int lane = tid & 31;
    const int lrow = lane & 15;
    const int sel  = lane >> 4;      // half-wave select
    const int warp = tid >> 5;
    const int wm   = warp & 1;       // 64-row slice
    const int wn   = warp >> 1;      // 64-col slice (0..3)

    const int n0 = blockIdx.x * BN;
    const int m0 = blockIdx.y * BM;
    const int batch = m0 >> 11;

    const unsigned smem_base = (unsigned)(unsigned long long)(const void*)smem;

    // ---- staging addresses, fully hoisted --------------------------------
    // A: 512 chunks (2/thread), rows tid>>2 and +64.  B: 1024 chunks (4/thread),
    // rows tid>>2, +64, +128, +192.  chunk col = tid&3 (16B within 64B row).
    const int srow = tid >> 2;
    const int scol = tid & 3;
    const unsigned ldsA0 = smem_base + srow * LDS_ROW_BYTES + scol * 16;
    const unsigned ldsA1 = ldsA0 + 64 * LDS_ROW_BYTES;
    const unsigned ldsBb = smem_base + AT_BYTES + srow * LDS_ROW_BYTES + scol * 16;
    const unsigned long long gA0 =
        (unsigned long long)(xb + (size_t)(m0 + srow) * K_TOTAL + scol * 8);
    const unsigned long long gA1 = gA0 + 2ull * 64 * K_TOTAL;
    const unsigned long long gB0 =
        (unsigned long long)(wb + (size_t)(n0 + srow) * K_TOTAL + scol * 8);

    v8f acc[4][4] = {};

    // prologue copy into buffer 0
    async_copy_b128(ldsA0, gA0);
    async_copy_b128(ldsA1, gA1);
#pragma unroll
    for (int j = 0; j < 4; ++j)
        async_copy_b128(ldsBb + j * (64 * LDS_ROW_BYTES),
                        gB0 + 2ull * (size_t)(j * 64) * K_TOTAL);

    const int nIter = K_TOTAL / BK;  // 128
#pragma unroll 1
    for (int it = 0; it < nIter; ++it) {
        wait_async0();
        __syncthreads();

        if (it + 1 < nIter) {
            const unsigned pOff = (unsigned)(((it + 1) & 1) * BUF_BYTES);
            const unsigned long long kOff = (unsigned long long)((it + 1) * BK) * 2u;
            async_copy_b128(ldsA0 + pOff, gA0 + kOff);
            async_copy_b128(ldsA1 + pOff, gA1 + kOff);
#pragma unroll
            for (int j = 0; j < 4; ++j)
                async_copy_b128(ldsBb + pOff + j * (64 * LDS_ROW_BYTES),
                                gB0 + kOff + 2ull * (size_t)(j * 64) * K_TOTAL);
        }

        const unsigned char* As = smem + (it & 1) * BUF_BYTES;
        const unsigned char* Bs = As + AT_BYTES;

        // B fragments (shared by all 4 mt): B stored [n][k]; lane = col n,
        // K sel*8..sel*8+7 and 16+sel*8..16+sel*8+7 (two 16B chunks).
        FragBf b[4];
#pragma unroll
        for (int nt = 0; nt < 4; ++nt) {
            const unsigned char* rp =
                Bs + (wn * 64 + nt * 16 + lrow) * LDS_ROW_BYTES + sel * 16;
            b[nt].q[0] = *(const uint4*)(rp);
            b[nt].q[1] = *(const uint4*)(rp + 32);
        }
        // stream A fragments, 4 WMMAs each (short live range)
#pragma unroll
        for (int mt = 0; mt < 4; ++mt) {
            FragBf a;
            const unsigned char* rp =
                As + (wm * 64 + mt * 16 + lrow) * LDS_ROW_BYTES + sel * 16;
            a.q[0] = *(const uint4*)(rp);        // K sel*8 .. sel*8+7
            a.q[1] = *(const uint4*)(rp + 32);   // K 16+sel*8 ..
#pragma unroll
            for (int nt = 0; nt < 4; ++nt)
                acc[mt][nt] = __builtin_amdgcn_wmma_f32_16x16x32_bf16(
                    false, a.v, false, b[nt].v, (short)0, acc[mt][nt], false, false);
        }
    }

    // ---- epilogue: out = acc + bias + SCALE * low @ loraB ----
    // C layout: vgpr i, lanes 0..15 -> M=i, N=lane ; lanes 16..31 -> M=8+i
    const float* lB = loraB + (size_t)batch * 16 * N_TOTAL;
#pragma unroll
    for (int nt = 0; nt < 4; ++nt) {
        const int n = n0 + wn * 64 + nt * 16 + lrow;
        const float bv = bias[n];
        float lb[16];
#pragma unroll
        for (int r = 0; r < 16; ++r) lb[r] = lB[(size_t)r * N_TOTAL + n];
#pragma unroll
        for (int mt = 0; mt < 4; ++mt) {
#pragma unroll
            for (int i = 0; i < 8; ++i) {
                const int m = m0 + wm * 64 + mt * 16 + sel * 8 + i;
                const float* lw = low + (size_t)m * 16;
                float s = 0.0f;
#pragma unroll
                for (int r = 0; r < 16; ++r) s = fmaf(lw[r], lb[r], s);
                out[(size_t)m * N_TOTAL + n] = acc[mt][nt][i] + bv + SCALE * s;
            }
        }
    }
}

// ---------------------------------------------------------------------------
extern "C" void kernel_launch(void* const* d_in, const int* in_sizes, int n_in,
                              void* d_out, int out_size, void* d_ws, size_t ws_size,
                              hipStream_t stream) {
    (void)in_sizes; (void)n_in; (void)out_size; (void)ws_size;
    const float* x      = (const float*)d_in[0];  // [4,2048,4096]
    const float* weight = (const float*)d_in[1];  // [4096,4096]
    const float* bias   = (const float*)d_in[2];  // [4096]
    const float* lora_A = (const float*)d_in[3];  // [4,4096,16]
    const float* lora_B = (const float*)d_in[4];  // [4,16,4096]
    float* out = (float*)d_out;                   // [4,2048,4096]

    // workspace: xb (64 MiB bf16) | wb (32 MiB bf16) | low (512 KiB f32)
    __bf16* xb  = (__bf16*)d_ws;
    __bf16* wb  = xb + (size_t)M_TOTAL * K_TOTAL;
    float*  low = (float*)(wb + (size_t)N_TOTAL * K_TOTAL);

    cvt_f32_to_bf16<<<2048, 256, 0, stream>>>(x, xb, M_TOTAL * K_TOTAL);
    cvt_f32_to_bf16<<<1024, 256, 0, stream>>>(weight, wb, N_TOTAL * K_TOTAL);
    lora_low_kernel<<<(M_TOTAL * 16) / 256, 256, 0, stream>>>(x, lora_A, low);

    dim3 grid(N_TOTAL / BN, M_TOTAL / BM);  // (16, 64)
    lora_gemm_kernel<<<grid, 256, 0, stream>>>(xb, wb, bias, low, lora_B, out);
}
